// DNCCell_25752623907117
// MI455X (gfx1250) — matrix operand order repaired and checked by reading, not compiled, not run on hardware
//
#include <hip/hip_runtime.h>
#include <math.h>

// Problem sizes (fixed by the reference)
#define Bn    256
#define Mn    64
#define Un    1024
#define Hn    16
#define Dn    4
#define POSn  32
#define PROJn 1056
#define EPSn  1e-3f

typedef __attribute__((ext_vector_type(16))) __bf16 v16bf;
typedef __attribute__((ext_vector_type(8)))  float  v8f;

__device__ __forceinline__ __bf16 bfc(float f) { return (__bf16)f; }

// Pack 8 consecutive floats (two float4) into fragment elements [base, base+8)
__device__ __forceinline__ void pack8(v16bf& a, int base, float4 lo, float4 hi) {
  a[base + 0] = bfc(lo.x); a[base + 1] = bfc(lo.y);
  a[base + 2] = bfc(lo.z); a[base + 3] = bfc(lo.w);
  a[base + 4] = bfc(hi.x); a[base + 5] = bfc(hi.y);
  a[base + 6] = bfc(hi.z); a[base + 7] = bfc(hi.w);
}

__device__ __forceinline__ float4 f4add(float4 a, float4 b) {
  return make_float4(a.x + b.x, a.y + b.y, a.z + b.z, a.w + b.w);
}

// ---------------------------------------------------------------------------
// K0: mem = state + inputs  (vectorized)
__global__ void init_mem(const float4* __restrict__ inp,
                         const float4* __restrict__ st,
                         float4* __restrict__ mem, int n4) {
  int i = blockIdx.x * blockDim.x + threadIdx.x;
  if (i < n4) mem[i] = f4add(inp[i], st[i]);
}

// ---------------------------------------------------------------------------
// KT: Wt[n][k] = bf16(W[k][n])  -- weight transpose+quantize for WMMA B frags
__global__ void w2t(const float* __restrict__ W, __bf16* __restrict__ Wt,
                    int K, int N) {
  int idx = blockIdx.x * blockDim.x + threadIdx.x;
  if (idx < K * N) {
    int n = idx / K, k = idx - n * K;
    Wt[(size_t)n * K + k] = bfc(W[(size_t)k * N + n]);
  }
}

// ---------------------------------------------------------------------------
// K1: mean over slots: mean[b][u] = (1/M) sum_m mem[b][m][u]
__global__ void slot_mean(const float* __restrict__ mem, float* __restrict__ mean) {
  int b = blockIdx.x;
  const float* mb = mem + (size_t)b * Mn * Un;
  for (int u = threadIdx.x; u < Un; u += blockDim.x) {
    float s = 0.f;
    for (int m = 0; m < Mn; ++m) s += mb[m * Un + u];
    mean[b * Un + u] = s * (1.f / Mn);
  }
}

// ---------------------------------------------------------------------------
// K2: logits[b*M+m][h] = sum_k keys(b,m,k) * W[k][h] + bias[h]
// keys(b,m,k) = (k<U ? mem[b][m][k]+mean[b][k] : pos[m][k-U]) (+ x[b][k] if HASX)
// Wt is bf16 [16][PROJ]. One wave per 16-row tile; no branches in k-loops.
template <bool HASX>
__global__ void __launch_bounds__(128)
keys_logits(const float* __restrict__ mem, const float* __restrict__ mean,
            const float* __restrict__ pos,  const float* __restrict__ xadd,
            const __bf16* __restrict__ Wt,  const float* __restrict__ bias,
            float* __restrict__ out) {
  int wave = threadIdx.x >> 5;
  int tile = blockIdx.x * 4 + wave;
  int row0 = tile * 16;                 // 16-row tile stays inside one b
  int b    = row0 >> 6;
  int m0   = row0 & 63;
  int lane = threadIdx.x & 31;
  int half = lane >> 4, r = lane & 15, n = lane & 15;
  const float* memm  = mem + (size_t)b * Mn * Un + (size_t)(m0 + r) * Un;
  const float* meanb = mean + b * Un;
  const float* posm  = pos + (m0 + r) * POSn;
  const float* xb    = xadd + b * PROJn;     // only dereferenced if HASX
  const __bf16* Wtn  = Wt + (size_t)n * PROJn;

  v8f acc = {};
  // main part: k in [0, 1024)
  for (int k0 = 0; k0 < Un; k0 += 32) {
    int b1 = k0 + (half << 4);
    int b2 = b1 + 8;
    float4 a0 = f4add(*(const float4*)(memm + b1), *(const float4*)(meanb + b1));
    float4 a1 = f4add(*(const float4*)(memm + b1 + 4), *(const float4*)(meanb + b1 + 4));
    float4 a2 = f4add(*(const float4*)(memm + b2), *(const float4*)(meanb + b2));
    float4 a3 = f4add(*(const float4*)(memm + b2 + 4), *(const float4*)(meanb + b2 + 4));
    if (HASX) {
      a0 = f4add(a0, *(const float4*)(xb + b1));
      a1 = f4add(a1, *(const float4*)(xb + b1 + 4));
      a2 = f4add(a2, *(const float4*)(xb + b2));
      a3 = f4add(a3, *(const float4*)(xb + b2 + 4));
    }
    v16bf a;
    pack8(a, 0, a0, a1);
    pack8(a, 8, a2, a3);
    v16bf bm = *(const v16bf*)(Wtn + k0 + (half << 4));
    acc = __builtin_amdgcn_wmma_f32_16x16x32_bf16(false, a, false, bm,
                                                  (short)0, acc, false, false);
  }
  // pos tail: k in [1024, 1056)
  {
    int b1 = (half << 4);
    int b2 = b1 + 8;
    float4 a0 = *(const float4*)(posm + b1);
    float4 a1 = *(const float4*)(posm + b1 + 4);
    float4 a2 = *(const float4*)(posm + b2);
    float4 a3 = *(const float4*)(posm + b2 + 4);
    if (HASX) {
      a0 = f4add(a0, *(const float4*)(xb + Un + b1));
      a1 = f4add(a1, *(const float4*)(xb + Un + b1 + 4));
      a2 = f4add(a2, *(const float4*)(xb + Un + b2));
      a3 = f4add(a3, *(const float4*)(xb + Un + b2 + 4));
    }
    v16bf a;
    pack8(a, 0, a0, a1);
    pack8(a, 8, a2, a3);
    v16bf bm = *(const v16bf*)(Wtn + Un + (half << 4));
    acc = __builtin_amdgcn_wmma_f32_16x16x32_bf16(false, a, false, bm,
                                                  (short)0, acc, false, false);
  }
  float bv = bias[n];
#pragma unroll
  for (int g = 0; g < 8; ++g)           // C layout: row = g + 8*half, col = n
    out[(row0 + g + (half << 3)) * Hn + n] = acc[g] + bv;
}

// ---------------------------------------------------------------------------
// K3: softmax over the M=64 slots, per (b,h). In-place on [B,M,H].
__global__ void softmax_slots(float* __restrict__ L) {
  int b = blockIdx.x;
  int h = threadIdx.x;
  if (h >= Hn) return;
  float* Lb = L + b * Mn * Hn;
  float mx = -1e30f;
  for (int m = 0; m < Mn; ++m) mx = fmaxf(mx, Lb[m * Hn + h]);
  float s = 0.f;
  for (int m = 0; m < Mn; ++m) {
    float e = __expf(Lb[m * Hn + h] - mx);
    Lb[m * Hn + h] = e; s += e;
  }
  float inv = 1.f / s;
  for (int m = 0; m < Mn; ++m) Lb[m * Hn + h] *= inv;
}

// ---------------------------------------------------------------------------
// K4: head-blocked weighted read: out[b][u] = sum_m wts[b][m][u>>6] * mem[b][m][u]
__global__ void att_read(const float* __restrict__ wts, const float* __restrict__ mem,
                         float* __restrict__ out, int ostride, int ooff) {
  int b = blockIdx.x;
  const float* mb = mem + (size_t)b * Mn * Un;
  const float* wb = wts + b * Mn * Hn;
  for (int u = threadIdx.x; u < Un; u += blockDim.x) {
    int h = u >> 6;
    float s = 0.f;
    for (int m = 0; m < Mn; ++m) s += wb[m * Hn + h] * mb[m * Un + u];
    out[(size_t)b * ostride + ooff + u] = s;
  }
}

// ---------------------------------------------------------------------------
// K5: C = act(A[Mrows,K] @ W[K,N] + bias). Wt = bf16 [N][K] (pre-transposed).
// Each wave computes a 16x32 strip: one A fragment, two B fragments, 2 WMMAs.
__global__ void __launch_bounds__(128)
gemm_bias_act(const float* __restrict__ A, const __bf16* __restrict__ Wt,
              const float* __restrict__ bias, float* __restrict__ C,
              int K, int N, int act) {
  int wave = threadIdx.x >> 5;
  int col0 = (blockIdx.x * 4 + wave) * 32;   // N here is a multiple of 32
  if (col0 >= N) return;                     // uniform per wave
  int row0 = blockIdx.y * 16;
  int lane = threadIdx.x & 31;
  int half = lane >> 4, r = lane & 15, n = lane & 15;
  const float*  Ar  = A + (size_t)(row0 + r) * K;
  const __bf16* Wt0 = Wt + (size_t)(col0 + n) * K;
  const __bf16* Wt1 = Wt0 + (size_t)16 * K;

  v8f acc0 = {}, acc1 = {};
  for (int k0 = 0; k0 < K; k0 += 32) {
    int b1 = k0 + (half << 4);
    int b2 = b1 + 8;
    float4 a0 = *(const float4*)(Ar + b1);
    float4 a1 = *(const float4*)(Ar + b1 + 4);
    float4 a2 = *(const float4*)(Ar + b2);
    float4 a3 = *(const float4*)(Ar + b2 + 4);
    v16bf a;
    pack8(a, 0, a0, a1);
    pack8(a, 8, a2, a3);
    int bo = k0 + (half << 4);
    v16bf bm0 = *(const v16bf*)(Wt0 + bo);
    v16bf bm1 = *(const v16bf*)(Wt1 + bo);
    acc0 = __builtin_amdgcn_wmma_f32_16x16x32_bf16(false, a, false, bm0,
                                                   (short)0, acc0, false, false);
    acc1 = __builtin_amdgcn_wmma_f32_16x16x32_bf16(false, a, false, bm1,
                                                   (short)0, acc1, false, false);
  }
  float bv0 = bias[col0 + n];
  float bv1 = bias[col0 + 16 + n];
#pragma unroll
  for (int g = 0; g < 8; ++g) {
    int row = row0 + g + (half << 3);
    float v0 = acc0[g] + bv0;
    float v1 = acc1[g] + bv1;
    if (act) { v0 = fmaxf(v0, 0.f); v1 = fmaxf(v1, 0.f); }
    C[(size_t)row * N + col0 + n]      = v0;
    C[(size_t)row * N + col0 + 16 + n] = v1;
  }
}

// ---------------------------------------------------------------------------
// K6: LayerNorm over PROJ, in place.
__global__ void layernorm(float* __restrict__ X, const float* __restrict__ gamma,
                          const float* __restrict__ beta) {
  __shared__ float s1[256], s2[256];
  int b = blockIdx.x, t = threadIdx.x;
  float* xb = X + b * PROJn;
  float s = 0.f, q = 0.f;
  for (int i = t; i < PROJn; i += 256) { float v = xb[i]; s += v; q += v * v; }
  s1[t] = s; s2[t] = q;
  __syncthreads();
  for (int o = 128; o > 0; o >>= 1) {
    if (t < o) { s1[t] += s1[t + o]; s2[t] += s2[t + o]; }
    __syncthreads();
  }
  float mu  = s1[0] * (1.f / PROJn);
  float var = s2[0] * (1.f / PROJn) - mu * mu;
  float inv = rsqrtf(var + EPSn);
  for (int i = t; i < PROJn; i += 256)
    xb[i] = gamma[i] * (xb[i] - mu) * inv + beta[i];
}

// ---------------------------------------------------------------------------
// K8: mem = (1-ww)*mem + ww*nm, ww broadcast per head, nm = X[b][:U]
__global__ void update_mem(float* __restrict__ mem, const float* __restrict__ ww,
                           const float* __restrict__ X) {
  size_t i = ((size_t)blockIdx.x * blockDim.x + threadIdx.x) * 4;
  int b   = (int)(i >> 16);      // / (M*U)
  int rem = (int)(i & 65535);
  int m   = rem >> 10;
  int u   = rem & 1023;          // multiple of 4; h = u>>6 constant over the 4
  float w  = ww[(b * Mn + m) * Hn + (u >> 6)];
  float4 nm = *(const float4*)(X + b * PROJn + u);
  float4 mv = *(float4*)(mem + i);
  float om = 1.f - w;
  mv.x = om * mv.x + w * nm.x;
  mv.y = om * mv.y + w * nm.y;
  mv.z = om * mv.z + w * nm.z;
  mv.w = om * mv.w + w * nm.w;
  *(float4*)(mem + i) = mv;
}

// ---------------------------------------------------------------------------
extern "C" void kernel_launch(void* const* d_in, const int* in_sizes, int n_in,
                              void* d_out, int out_size, void* d_ws, size_t ws_size,
                              hipStream_t stream) {
  const float* inputs = (const float*)d_in[0];
  const float* state  = (const float*)d_in[1];
  const float* pos    = (const float*)d_in[2];
  const float* gamma  = (const float*)d_in[3];
  const float* beta   = (const float*)d_in[4];
  const float* readW  = (const float*)d_in[5];
  const float* readb  = (const float*)d_in[6];
  const float* writeW = (const float*)d_in[7];
  const float* writeb = (const float*)d_in[8];
  const float* kernW  = (const float*)d_in[9];
  const float* kernb  = (const float*)d_in[10];
  const float* projW  = (const float*)d_in[11];
  const float* projb  = (const float*)d_in[12];
  const float* r1W    = (const float*)d_in[13];
  const float* r1b    = (const float*)d_in[14];
  const float* r2W    = (const float*)d_in[15];
  const float* r2b    = (const float*)d_in[16];
  float* out = (float*)d_out;

  // Workspace carve-out: ~74.5 MB
  float* mem  = (float*)d_ws;                      // [B,M,U]
  float* mean = mem  + (size_t)Bn * Mn * Un;       // [B,U]
  float* wl   = mean + (size_t)Bn * Un;            // [B,M,H] logits -> weights
  float* att  = wl   + (size_t)Bn * Mn * Hn;       // [B,U]
  float* t1   = att  + (size_t)Bn * Un;            // [B,U]
  float* x    = t1   + (size_t)Bn * Un;            // [B,PROJ]
  __bf16* wtb = (__bf16*)(x + (size_t)Bn * PROJn); // up to [1056][1024] bf16

  const int memN4  = Bn * Mn * Un / 4;
  const int klBlks = Bn * Mn / 64;                 // 1024 row tiles / 4 waves
  const int smallT = (PROJn * Hn + 255) / 256;     // transpose blocks, small W
  const int kernT  = (Un * Un + 255) / 256;
  const int projT  = (Un * PROJn + 255) / 256;

  init_mem<<<memN4 / 256, 256, 0, stream>>>((const float4*)inputs,
                                            (const float4*)state,
                                            (float4*)mem, memN4);

  for (int l = 0; l < Dn; ++l) {
    slot_mean<<<Bn, 256, 0, stream>>>(mem, mean);

    w2t<<<smallT, 256, 0, stream>>>(readW + (size_t)l * PROJn * Hn, wtb, PROJn, Hn);
    keys_logits<false><<<klBlks, 128, 0, stream>>>(mem, mean, pos, nullptr,
                                                   wtb, readb + l * Hn, wl);
    softmax_slots<<<Bn, 32, 0, stream>>>(wl);
    att_read<<<Bn, 256, 0, stream>>>(wl, mem, att, Un, 0);

    w2t<<<kernT, 256, 0, stream>>>(kernW + (size_t)l * Un * Un, wtb, Un, Un);
    gemm_bias_act<<<dim3(Un / 128, Bn / 16), 128, 0, stream>>>(
        att, wtb, kernb + l * Un, t1, Un, Un, 1);

    w2t<<<projT, 256, 0, stream>>>(projW + (size_t)l * Un * PROJn, wtb, Un, PROJn);
    gemm_bias_act<<<dim3((PROJn + 127) / 128, Bn / 16), 128, 0, stream>>>(
        t1, wtb, projb + l * PROJn, x, Un, PROJn, 0);

    layernorm<<<Bn, 256, 0, stream>>>(x, gamma, beta);

    w2t<<<smallT, 256, 0, stream>>>(writeW + (size_t)l * PROJn * Hn, wtb, PROJn, Hn);
    keys_logits<true><<<klBlks, 128, 0, stream>>>(mem, mean, pos, x,
                                                  wtb, writeb + l * Hn, wl);
    softmax_slots<<<Bn, 32, 0, stream>>>(wl);
    update_mem<<<memN4 / 256, 256, 0, stream>>>(mem, wl, x);
  }

  // Final two reads into out[:, :2048]
  slot_mean<<<Bn, 256, 0, stream>>>(mem, mean);
  w2t<<<smallT, 256, 0, stream>>>(r1W, wtb, PROJn, Hn);
  keys_logits<false><<<klBlks, 128, 0, stream>>>(mem, mean, pos, nullptr,
                                                 wtb, r1b, wl);
  softmax_slots<<<Bn, 32, 0, stream>>>(wl);
  att_read<<<Bn, 256, 0, stream>>>(wl, mem, out, 2 * Un, 0);

  w2t<<<smallT, 256, 0, stream>>>(r2W, wtb, PROJn, Hn);
  keys_logits<false><<<klBlks, 128, 0, stream>>>(mem, mean, pos, nullptr,
                                                 wtb, r2b, wl);
  softmax_slots<<<Bn, 32, 0, stream>>>(wl);
  att_read<<<Bn, 256, 0, stream>>>(wl, mem, out, 2 * Un, Un);

  // Second tuple element: flattened mem
  hipMemcpyAsync(out + (size_t)Bn * 2 * Un, mem,
                 (size_t)Bn * Mn * Un * sizeof(float),
                 hipMemcpyDeviceToDevice, stream);
}
